// Any4Linear_2997887172647
// MI455X (gfx1250) — matrix operand order repaired and checked by reading, not compiled, not run on hardware
//
#include <hip/hip_runtime.h>

// ---- CDNA5 WMMA types (probe-verified signatures) ----
typedef __attribute__((ext_vector_type(16))) __bf16        v16bf;
typedef __attribute__((ext_vector_type(8)))  float         v8f;
typedef __attribute__((ext_vector_type(8)))  unsigned int  v8u;
typedef __attribute__((ext_vector_type(4)))  int           v4i;

#define BK 32
#define LDSS 40   // bf16 per LDS row: 32 data + 8 pad (80B: 16B-aligned, stride 20 words)
// fused-fallback tile
#define BM 128
#define BN 128
// main GEMM tile
#define GM 128
#define GN 256

__device__ __forceinline__ unsigned short f2bf(float f) {
    unsigned int u = __float_as_uint(f);
    u += 0x7FFFu + ((u >> 16) & 1u);
    return (unsigned short)(u >> 16);
}

// ---- async global->LDS copy (CDNA5 ASYNCcnt path), with safe fallback ----
__device__ __forceinline__ void async_cp16(const void* g, unsigned short* l) {
#if __has_builtin(__builtin_amdgcn_global_load_async_to_lds_b128)
    __builtin_amdgcn_global_load_async_to_lds_b128(
        (__attribute__((address_space(1))) v4i*)g,
        (__attribute__((address_space(3))) v4i*)l, 0, 0);
#else
    *(uint4*)l = *(const uint4*)g;
#endif
}

__device__ __forceinline__ void wait_async0() {
#if __has_builtin(__builtin_amdgcn_s_wait_asynccnt)
    __builtin_amdgcn_s_wait_asynccnt(0);
#else
    asm volatile("s_wait_asynccnt 0x0" ::: "memory");
#endif
}

// ============================================================
// Prep kernel 1: fp32 -> bf16 (8 elements / thread)
// ============================================================
__global__ __launch_bounds__(256)
void cvt_bf16_kernel(const float* __restrict__ src, unsigned short* __restrict__ dst, long n8) {
    long i = (long)blockIdx.x * 256 + threadIdx.x;
    if (i >= n8) return;
    const float4 a = ((const float4*)src)[2 * i];
    const float4 b = ((const float4*)src)[2 * i + 1];
    uint4 o;
    o.x = (unsigned int)f2bf(a.x) | ((unsigned int)f2bf(a.y) << 16);
    o.y = (unsigned int)f2bf(a.z) | ((unsigned int)f2bf(a.w) << 16);
    o.z = (unsigned int)f2bf(b.x) | ((unsigned int)f2bf(b.y) << 16);
    o.w = (unsigned int)f2bf(b.z) | ((unsigned int)f2bf(b.w) << 16);
    ((uint4*)dst)[i] = o;
}

// ============================================================
// Prep kernel 2: any4 LUT dequant -> bf16 (8 codes / thread)
// ============================================================
__global__ __launch_bounds__(256)
void deq_w_kernel(const int* __restrict__ Wc, const float* __restrict__ lut,
                  const float* __restrict__ snz, unsigned short* __restrict__ Wb,
                  int I, int O, int G) {
    long i = (long)blockIdx.x * 256 + threadIdx.x;
    long nchunks = (long)O * I / 8;
    if (i >= nchunks) return;
    long e0 = i * 8;
    int o = (int)(e0 / I);
    int k = (int)(e0 % I);
    int g = k / G;
    const float2 sz = *(const float2*)(snz + ((long)g * O + o) * 2);
    const float* lr = lut + (long)o * 16;
    const int4 c0 = *(const int4*)(Wc + e0);
    const int4 c1 = *(const int4*)(Wc + e0 + 4);
    float w0 = fmaf(lr[c0.x & 15], sz.x, sz.y);
    float w1 = fmaf(lr[c0.y & 15], sz.x, sz.y);
    float w2 = fmaf(lr[c0.z & 15], sz.x, sz.y);
    float w3 = fmaf(lr[c0.w & 15], sz.x, sz.y);
    float w4 = fmaf(lr[c1.x & 15], sz.x, sz.y);
    float w5 = fmaf(lr[c1.y & 15], sz.x, sz.y);
    float w6 = fmaf(lr[c1.z & 15], sz.x, sz.y);
    float w7 = fmaf(lr[c1.w & 15], sz.x, sz.y);
    uint4 out;
    out.x = (unsigned int)f2bf(w0) | ((unsigned int)f2bf(w1) << 16);
    out.y = (unsigned int)f2bf(w2) | ((unsigned int)f2bf(w3) << 16);
    out.z = (unsigned int)f2bf(w4) | ((unsigned int)f2bf(w5) << 16);
    out.w = (unsigned int)f2bf(w6) | ((unsigned int)f2bf(w7) << 16);
    ((uint4*)(Wb + e0))[0] = out;
}

// ============================================================
// Main GEMM: bf16 A/B, f32 accum, async double-buffered LDS
// Block 128(M) x 256(N); 8 waves in 2(M) x 4(N); wave tile 64x64 = 4x4 WMMA
// 16 ds_load_b128 : 16 WMMA per wave per K-step (1:1)
// ============================================================
__global__ __launch_bounds__(256, 1)
void gemm_bf16_wmma(const unsigned short* __restrict__ A,   // [M,I] bf16
                    const unsigned short* __restrict__ B,   // [O,I] bf16
                    const float* __restrict__ bias,
                    float* __restrict__ Y,
                    int M, int I, int O)
{
    __shared__ __align__(16) unsigned short lA[2][GM * LDSS];   // 2 x 10240 B
    __shared__ __align__(16) unsigned short lB[2][GN * LDSS];   // 2 x 20480 B

    const int tid  = threadIdx.x;
    const int lane = tid & 31;
    const int wv   = tid >> 5;
    const int wm   = wv & 1;        // 2 waves x 64 rows (M)
    const int wn   = wv >> 1;       // 4 waves x 64 cols (N)
    const int half = lane >> 4;
    const int l16  = lane & 15;

    const long blockN = (long)blockIdx.x * GN;
    const long blockM = (long)blockIdx.y * GM;

    // staging: A tile 128x32 (512 x 16B chunks -> 2/thread), B tile 256x32 (1024 -> 4/thread)
    const int srow = tid >> 2;             // 0..63
    const int scol = (tid & 3) * 8;        // bf16 offset in row
    const unsigned short* gA0 = A + (blockM + srow)       * (long)I + scol;
    const unsigned short* gA1 = A + (blockM + srow + 64)  * (long)I + scol;
    const unsigned short* gB0 = B + (blockN + srow)       * (long)I + scol;
    const unsigned short* gB1 = B + (blockN + srow + 64)  * (long)I + scol;
    const unsigned short* gB2 = B + (blockN + srow + 128) * (long)I + scol;
    const unsigned short* gB3 = B + (blockN + srow + 192) * (long)I + scol;
    const int loA0 = srow * LDSS + scol;
    const int loA1 = (srow + 64) * LDSS + scol;
    const int loB1 = (srow + 64) * LDSS + scol;
    const int loB2 = (srow + 128) * LDSS + scol;
    const int loB3 = (srow + 192) * LDSS + scol;

    auto stage = [&](int buf, int kb) {
        async_cp16(gA0 + kb, &lA[buf][loA0]);
        async_cp16(gA1 + kb, &lA[buf][loA1]);
        async_cp16(gB0 + kb, &lB[buf][loA0]);
        async_cp16(gB1 + kb, &lB[buf][loB1]);
        async_cp16(gB2 + kb, &lB[buf][loB2]);
        async_cp16(gB3 + kb, &lB[buf][loB3]);
    };

    v8f acc[4][4];
#pragma unroll
    for (int tm = 0; tm < 4; ++tm)
#pragma unroll
        for (int tn = 0; tn < 4; ++tn)
            acc[tm][tn] = v8f{};

    const int nK = I / BK;
    stage(0, 0);

    for (int kb = 0; kb < nK; ++kb) {
        const int cur = kb & 1;
        wait_async0();            // my wave's loads for `cur` complete
        __syncthreads();          // all waves' loads visible; prev reads finished
        if (kb + 1 < nK) stage(cur ^ 1, (kb + 1) * BK);   // overlap next DMA with compute

        // ---- fragments per ISA VGPR layouts ----
        v16bf afrag[4], bfrag[4];
#pragma unroll
        for (int tm = 0; tm < 4; ++tm) {
            // A 16x32 bf16: lanes 0-15 K 0..7 & 16..23 ; lanes 16-31 K 8..15 & 24..31
            const unsigned short* pa = &lA[cur][(wm * 64 + tm * 16 + l16) * LDSS + half * 8];
            uint4 lo = *(const uint4*)pa;
            uint4 hi = *(const uint4*)(pa + 16);
            v8u t;
            t[0] = lo.x; t[1] = lo.y; t[2] = lo.z; t[3] = lo.w;
            t[4] = hi.x; t[5] = hi.y; t[6] = hi.z; t[7] = hi.w;
            afrag[tm] = __builtin_bit_cast(v16bf, t);
        }
#pragma unroll
        for (int tn = 0; tn < 4; ++tn) {
            // B 32x16 bf16: lane n = col; lanes 0-15 K 0..7 & 16..23 ; lanes 16-31 K 8..15 & 24..31
            const unsigned short* pb = &lB[cur][(wn * 64 + tn * 16 + l16) * LDSS + half * 8];
            uint4 lo = *(const uint4*)pb;
            uint4 hi = *(const uint4*)(pb + 16);
            v8u t;
            t[0] = lo.x; t[1] = lo.y; t[2] = lo.z; t[3] = lo.w;
            t[4] = hi.x; t[5] = hi.y; t[6] = hi.z; t[7] = hi.w;
            bfrag[tn] = __builtin_bit_cast(v16bf, t);
        }

#pragma unroll
        for (int tm = 0; tm < 4; ++tm)
#pragma unroll
            for (int tn = 0; tn < 4; ++tn)
                acc[tm][tn] = __builtin_amdgcn_wmma_f32_16x16x32_bf16(
                    false, afrag[tm], false, bfrag[tn],
                    (short)0, acc[tm][tn], false, false);
    }

    // ---- epilogue: pointer-increment stores, bias fused ----
#pragma unroll
    for (int tm = 0; tm < 4; ++tm) {
#pragma unroll
        for (int tn = 0; tn < 4; ++tn) {
            const long n = blockN + wn * 64 + tn * 16 + l16;
            const float bv = bias[n];
            float* yp = Y + (size_t)(blockM + wm * 64 + tm * 16 + half * 8) * O + n;
#pragma unroll
            for (int r = 0; r < 8; ++r) {
                *yp = acc[tm][tn][r] + bv;
                yp += O;
            }
        }
    }
}

// ============================================================
// Fallback: fused dequant GEMM (used if d_ws too small or O % GN / M % GM nonzero)
// ============================================================
__global__ __launch_bounds__(256, 1)
void any4_wmma_fused(const float* __restrict__ X, const int* __restrict__ Wc,
                     const float* __restrict__ lut, const float* __restrict__ snz,
                     const float* __restrict__ bias, float* __restrict__ Y,
                     int M, int I, int O, int G)
{
    __shared__ __align__(16) unsigned short sA[BM * LDSS];
    __shared__ __align__(16) unsigned short sB[BN * LDSS];

    const int tid  = threadIdx.x;
    const int lane = tid & 31;
    const int wv   = tid >> 5;
    const int wm   = wv & 3;
    const int wn   = wv >> 2;
    const int half = lane >> 4;
    const int l16  = lane & 15;
    const long blockN = (long)blockIdx.x * BN;
    const long blockM = (long)blockIdx.y * BM;

    v8f acc[2][4];
#pragma unroll
    for (int tm = 0; tm < 2; ++tm)
#pragma unroll
        for (int tn = 0; tn < 4; ++tn)
            acc[tm][tn] = v8f{};

    for (int kb = 0; kb < I; kb += BK) {
#pragma unroll
        for (int s = 0; s < 4; ++s) {
            int idx = tid + s * 256;
            int row = idx >> 3;
            int kc  = (idx & 7) << 2;
            const float4 v = *(const float4*)(X + (blockM + row) * (long)I + kb + kc);
            unsigned int p0 = (unsigned int)f2bf(v.x) | ((unsigned int)f2bf(v.y) << 16);
            unsigned int p1 = (unsigned int)f2bf(v.z) | ((unsigned int)f2bf(v.w) << 16);
            *(uint2*)&sA[row * LDSS + kc] = make_uint2(p0, p1);
        }
#pragma unroll
        for (int s = 0; s < 4; ++s) {
            int idx = tid + s * 256;
            int row = idx >> 3;
            int kc  = (idx & 7) << 2;
            long o  = blockN + row;
            const int4 c = *(const int4*)(Wc + o * (long)I + kb + kc);
            int g = (kb + kc) / G;
            const float2 sz = *(const float2*)(snz + ((long)g * O + o) * 2);
            const float* lr = lut + o * 16;
            float w0 = fmaf(lr[c.x & 15], sz.x, sz.y);
            float w1 = fmaf(lr[c.y & 15], sz.x, sz.y);
            float w2 = fmaf(lr[c.z & 15], sz.x, sz.y);
            float w3 = fmaf(lr[c.w & 15], sz.x, sz.y);
            unsigned int p0 = (unsigned int)f2bf(w0) | ((unsigned int)f2bf(w1) << 16);
            unsigned int p1 = (unsigned int)f2bf(w2) | ((unsigned int)f2bf(w3) << 16);
            *(uint2*)&sB[row * LDSS + kc] = make_uint2(p0, p1);
        }
        __syncthreads();

        v16bf afrag[2], bfrag[4];
#pragma unroll
        for (int tm = 0; tm < 2; ++tm) {
            const unsigned short* pa = &sA[(wm * 32 + tm * 16 + l16) * LDSS + half * 8];
            uint4 lo = *(const uint4*)pa;
            uint4 hi = *(const uint4*)(pa + 16);
            v8u t;
            t[0] = lo.x; t[1] = lo.y; t[2] = lo.z; t[3] = lo.w;
            t[4] = hi.x; t[5] = hi.y; t[6] = hi.z; t[7] = hi.w;
            afrag[tm] = __builtin_bit_cast(v16bf, t);
        }
#pragma unroll
        for (int tn = 0; tn < 4; ++tn) {
            const unsigned short* pb = &sB[(wn * 64 + tn * 16 + l16) * LDSS + half * 8];
            uint4 lo = *(const uint4*)pb;
            uint4 hi = *(const uint4*)(pb + 16);
            v8u t;
            t[0] = lo.x; t[1] = lo.y; t[2] = lo.z; t[3] = lo.w;
            t[4] = hi.x; t[5] = hi.y; t[6] = hi.z; t[7] = hi.w;
            bfrag[tn] = __builtin_bit_cast(v16bf, t);
        }
#pragma unroll
        for (int tm = 0; tm < 2; ++tm)
#pragma unroll
            for (int tn = 0; tn < 4; ++tn)
                acc[tm][tn] = __builtin_amdgcn_wmma_f32_16x16x32_bf16(
                    false, afrag[tm], false, bfrag[tn],
                    (short)0, acc[tm][tn], false, false);
        __syncthreads();
    }

#pragma unroll
    for (int tm = 0; tm < 2; ++tm) {
#pragma unroll
        for (int tn = 0; tn < 4; ++tn) {
            const long n = blockN + wn * 64 + tn * 16 + l16;
            const float bv = bias[n];
            float* yp = Y + (size_t)(blockM + wm * 32 + tm * 16 + half * 8) * O + n;
#pragma unroll
            for (int r = 0; r < 8; ++r) {
                *yp = acc[tm][tn][r] + bv;
                yp += O;
            }
        }
    }
}

extern "C" void kernel_launch(void* const* d_in, const int* in_sizes, int n_in,
                              void* d_out, int out_size, void* d_ws, size_t ws_size,
                              hipStream_t stream) {
    const float* X    = (const float*)d_in[0];
    const int*   Wc   = (const int*)d_in[1];
    const float* lut  = (const float*)d_in[2];
    const float* snz  = (const float*)d_in[3];
    const float* bias = (const float*)d_in[4];
    float* Y = (float*)d_out;

    const int O = in_sizes[2] / 16;
    const int I = in_sizes[1] / O;
    const int M = in_sizes[0] / I;
    const int nG = in_sizes[3] / (O * 2);
    const int G = I / nG;

    const size_t needW = (size_t)O * I * 2;   // bf16 W
    const size_t needX = (size_t)M * I * 2;   // bf16 X
    dim3 block(256);

    if (ws_size >= needW + needX && (O % GN) == 0 && (M % GM) == 0) {
        unsigned short* Wb = (unsigned short*)d_ws;
        unsigned short* Xb = Wb + (size_t)O * I;

        const long wChunks = (long)O * I / 8;
        deq_w_kernel<<<dim3((wChunks + 255) / 256), block, 0, stream>>>(Wc, lut, snz, Wb, I, O, G);

        const long xChunks = (long)M * I / 8;
        cvt_bf16_kernel<<<dim3((xChunks + 255) / 256), block, 0, stream>>>(X, Xb, xChunks);

        dim3 grid(O / GN, M / GM);
        gemm_bf16_wmma<<<grid, block, 0, stream>>>(Xb, Wb, bias, Y, M, I, O);
    } else {
        dim3 grid(O / BN, M / BM);
        any4_wmma_fused<<<grid, block, 0, stream>>>(X, Wc, lut, snz, bias, Y, M, I, O, G);
    }
}
